// Yolov3_Output_Extractor_63445256897064
// MI455X (gfx1250) — 8-bit matrix operand order repaired and checked by reading, compiled, not run on hardware
//
#include <hip/hip_runtime.h>
#include <stdint.h>

#define BATCH 16
#define NANCH 22743
#define NCLS  80
#define ROWF  85
#define TOPK  1000
#define KPAD  1024
#define CONF_T 0.5f
#define NMS_T  0.4f

typedef int v8i __attribute__((ext_vector_type(8)));

__device__ __forceinline__ unsigned mono32(float f) {
  unsigned u = __float_as_uint(f);
  return (u & 0x80000000u) ? ~u : (u | 0x80000000u);
}

__device__ __forceinline__ unsigned onehotw(int cls, int kb) {
  unsigned d = (unsigned)(cls - kb);
  return (d < 4u) ? (1u << (d * 8u)) : 0u;
}

__global__ void score_keys_kernel(const float* __restrict__ v3,
                                  unsigned long long* __restrict__ keys) {
  int gi = blockIdx.x * blockDim.x + threadIdx.x;
  if (gi >= BATCH * NANCH) return;
  const float* row = v3 + (size_t)gi * ROWF;
  __builtin_prefetch(row + ROWF * 512, 0, 0);
  float obj = row[4];
  float best = row[5];
#pragma unroll 8
  for (int k = 1; k < NCLS; ++k) best = fmaxf(best, row[5 + k]);
  float score = (obj >= CONF_T) ? obj * best : -1.0f;
  unsigned n = (unsigned)(gi % NANCH);
  keys[gi] = ((unsigned long long)mono32(score) << 32) |
             (unsigned long long)(0xFFFFFFFFu - n);
}

__global__ void select_sort_kernel(const unsigned long long* __restrict__ keys,
                                   unsigned long long* __restrict__ topkeys) {
  __shared__ unsigned hist[256];
  __shared__ unsigned long long sPrefix;
  __shared__ unsigned sRank;
  __shared__ unsigned sCnt;
  __shared__ unsigned long long topk[KPAD];

  const int b   = blockIdx.x;
  const int tid = threadIdx.x;
  const unsigned long long* kimg = keys + (size_t)b * NANCH;

  unsigned long long prefix = 0ull;
  unsigned rank = TOPK;

  for (int d = 7; d >= 0; --d) {
    hist[tid] = 0;
    __syncthreads();
    for (int n = tid; n < NANCH; n += 256) {
      unsigned long long k = kimg[n];
      bool match = (d == 7) || ((k >> (unsigned)((d + 1) * 8)) == prefix);
      if (match) atomicAdd(&hist[(unsigned)(k >> (unsigned)(d * 8)) & 255u], 1u);
    }
    __syncthreads();
    if (tid == 0) {
      unsigned cum = 0; int v = 255;
      for (; v >= 0; --v) { cum += hist[v]; if (cum >= rank) break; }
      if (v < 0) v = 0;
      sRank = rank - (cum - hist[v]);
      sPrefix = (prefix << 8) | (unsigned long long)(unsigned)v;
    }
    __syncthreads();
    prefix = sPrefix; rank = sRank;
    __syncthreads();
  }
  const unsigned long long thresh = prefix;

  for (int t = tid; t < KPAD; t += 256) topk[t] = 0ull;
  if (tid == 0) sCnt = 0;
  __syncthreads();
  for (int n = tid; n < NANCH; n += 256) {
    unsigned long long k = kimg[n];
    if (k >= thresh) {
      unsigned pos = atomicAdd(&sCnt, 1u);
      if (pos < KPAD) topk[pos] = k;
    }
  }
  __syncthreads();

  for (unsigned sz = 2; sz <= KPAD; sz <<= 1) {
    for (unsigned st = sz >> 1; st > 0; st >>= 1) {
      for (unsigned i = tid; i < KPAD; i += 256) {
        unsigned l = i ^ st;
        if (l > i) {
          unsigned long long a = topk[i], c = topk[l];
          bool sw = ((i & sz) == 0) ? (a < c) : (a > c);
          if (sw) { topk[i] = c; topk[l] = a; }
        }
      }
      __syncthreads();
    }
  }
  for (int t = tid; t < KPAD; t += 256)
    topkeys[(size_t)b * KPAD + t] = topk[t];
}

__global__ void gather_kernel(const float* __restrict__ v3,
                              const unsigned long long* __restrict__ topkeys,
                              float* __restrict__ fx1, float* __restrict__ fy1,
                              float* __restrict__ fx2, float* __restrict__ fy2,
                              float* __restrict__ farea,
                              float* __restrict__ fobj, float* __restrict__ fconf,
                              int* __restrict__ fcls,
                              unsigned* __restrict__ validmask) {
  const int b = blockIdx.x;
  const int tid = threadIdx.x;
  for (int r = tid; r < KPAD; r += 256) {
    unsigned long long k = topkeys[(size_t)b * KPAD + r];
    unsigned idx = 0xFFFFFFFFu - (unsigned)(k & 0xFFFFFFFFull);
    unsigned mhi = (unsigned)(k >> 32);
    bool valid = (r < TOPK) && (mhi > 0x80000000u) && (idx < NANCH);
    unsigned si = (idx < NANCH) ? idx : 0u;
    const float* row = v3 + ((size_t)b * NANCH + si) * ROWF;
    float cx = row[0], cy = row[1], w = row[2], h = row[3], obj = row[4];
    float x1 = cx - w * 0.5f, y1 = cy - h * 0.5f;
    float x2 = cx + w * 0.5f, y2 = cy + h * 0.5f;
    float best = row[5]; int cid = 0;
    for (int c = 1; c < NCLS; ++c) {
      float s = row[5 + c];
      if (s > best) { best = s; cid = c; }
    }
    float area = fmaxf(x2 - x1, 0.0f) * fmaxf(y2 - y1, 0.0f);
    int o = b * KPAD + r;
    fx1[o] = x1; fy1[o] = y1; fx2[o] = x2; fy2[o] = y2;
    farea[o] = area; fobj[o] = obj; fconf[o] = best; fcls[o] = cid;
    unsigned bal = (unsigned)__ballot(valid);
    if ((tid & 31) == 0) validmask[b * 32 + (r >> 5)] = bal;
  }
}

__global__ void suppress_kernel(const float* __restrict__ fx1, const float* __restrict__ fy1,
                                const float* __restrict__ fx2, const float* __restrict__ fy2,
                                const float* __restrict__ farea, const int* __restrict__ fcls,
                                unsigned* __restrict__ sup) {
  __shared__ float sX1[KPAD], sY1[KPAD], sX2[KPAD], sY2[KPAD], sA[KPAD];
  __shared__ int   sC[KPAD];

  const int b  = blockIdx.x >> 6;
  const int ri = blockIdx.x & 63;
  const int i0 = ri * 16;
  const int tid = threadIdx.x;

  for (int t = tid; t < KPAD; t += 256) {
    int o = b * KPAD + t;
    sX1[t] = fx1[o]; sY1[t] = fy1[o]; sX2[t] = fx2[o]; sY2[t] = fy2[o];
    sA[t] = farea[o]; sC[t] = fcls[o];
  }
  __syncthreads();

  const int lane = tid & 31;
  const int wv   = tid >> 5;
  const int h    = lane >> 4;
  const int n15  = lane & 15;

  int clsRow = sC[i0 + n15];
  v8i A0 = {0,0,0,0,0,0,0,0}, A1 = {0,0,0,0,0,0,0,0};
#pragma unroll
  for (int v = 0; v < 8; ++v) {
    int kb = (v >> 1) * 16 + h * 8 + (v & 1) * 4;
    A0[v] = (int)onehotw(clsRow, kb);
    A1[v] = (int)onehotw(clsRow, kb + 64);
  }

  unsigned* rowbase = sup + (size_t)b * KPAD * 32;

  for (int word = wv; word < 32; word += 8) {
    unsigned bal0[8], bal1[8];
#pragma unroll
    for (int t = 0; t < 2; ++t) {
      const int j0 = word * 32 + t * 16;
      int clsCol = sC[j0 + n15];
      v8i B0 = {0,0,0,0,0,0,0,0}, B1 = {0,0,0,0,0,0,0,0};
#pragma unroll
      for (int v = 0; v < 8; ++v) {
        int kb = (v >> 1) * 16 + h * 8 + (v & 1) * 4;
        B0[v] = (int)onehotw(clsCol, kb);
        B1[v] = (int)onehotw(clsCol, kb + 64);
      }
      v8i cz = {0,0,0,0,0,0,0,0};
      v8i d = __builtin_amdgcn_wmma_i32_16x16x64_iu8(false, A0, false, B0, cz, false, false);
      d     = __builtin_amdgcn_wmma_i32_16x16x64_iu8(false, A1, false, B1, d,  false, false);

      const int j = j0 + n15;
      float jx1 = sX1[j], jy1 = sY1[j], jx2 = sX2[j], jy2 = sY2[j], ja = sA[j];
#pragma unroll
      for (int v = 0; v < 8; ++v) {
        int i = i0 + v + h * 8;
        float xx1 = fmaxf(sX1[i], jx1), yy1 = fmaxf(sY1[i], jy1);
        float xx2 = fminf(sX2[i], jx2), yy2 = fminf(sY2[i], jy2);
        float inter = fmaxf(xx2 - xx1, 0.0f) * fmaxf(yy2 - yy1, 0.0f);
        float iou = inter / (sA[i] + ja - inter + 1e-16f);
        bool pred = (d[v] != 0) && (iou > NMS_T) && (j > i) && (j < TOPK);
        unsigned bal = (unsigned)__ballot(pred);
        if (t == 0) bal0[v] = bal; else bal1[v] = bal;
      }
    }
#pragma unroll
    for (int v = 0; v < 8; ++v) {
      unsigned wlo = (bal0[v] & 0xFFFFu) | (bal1[v] << 16);
      unsigned whi = (bal0[v] >> 16)     | (bal1[v] & 0xFFFF0000u);
      if (lane == v)     rowbase[(size_t)(i0 + v)     * 32 + word] = wlo;
      if (lane == v + 8) rowbase[(size_t)(i0 + v + 8) * 32 + word] = whi;
    }
  }
}

__global__ void nms_out_kernel(const unsigned* __restrict__ sup,
                               const unsigned* __restrict__ validmask,
                               const float* __restrict__ fx1, const float* __restrict__ fy1,
                               const float* __restrict__ fx2, const float* __restrict__ fy2,
                               const float* __restrict__ fobj, const float* __restrict__ fconf,
                               const int* __restrict__ fcls,
                               float* __restrict__ out) {
  extern __shared__ unsigned smem[];
  unsigned* supLds  = smem;
  unsigned* keepLds = smem + TOPK * 32;

  const int b = blockIdx.x;
  const int tid = threadIdx.x;
  const unsigned* sb = sup + (size_t)b * KPAD * 32;

  for (int t = tid; t < TOPK * 32; t += 1024) supLds[t] = sb[t];
  if (tid < 32) keepLds[tid] = validmask[b * 32 + tid];
  __syncthreads();

  if (tid < 32) {
    unsigned my = keepLds[tid];
    for (int i = 0; i < TOPK; ++i) {
      unsigned w = (unsigned)__shfl((int)my, i >> 5, 32);
      if (w & (1u << (i & 31))) my &= ~supLds[i * 32 + tid];
    }
    keepLds[tid] = my;
  }
  __syncthreads();

  if (tid < TOPK) {
    int r = tid;
    float m = ((keepLds[r >> 5] >> (r & 31)) & 1u) ? 1.0f : 0.0f;
    int o = b * KPAD + r;
    float* dst = out + ((size_t)b * TOPK + r) * 7;
    dst[0] = fx1[o] * m;  dst[1] = fy1[o] * m;
    dst[2] = fx2[o] * m;  dst[3] = fy2[o] * m;
    dst[4] = fobj[o] * m; dst[5] = fconf[o] * m;
    dst[6] = (float)fcls[o] * m;
  }
}

extern "C" void kernel_launch(void* const* d_in, const int* in_sizes, int n_in,
                              void* d_out, int out_size, void* d_ws, size_t ws_size,
                              hipStream_t stream) {
  (void)in_sizes; (void)n_in; (void)out_size; (void)ws_size;
  const float* v3 = (const float*)d_in[0];
  float* out = (float*)d_out;

  char* ws = (char*)d_ws;
  size_t off = 0;
  unsigned long long* wsKeys = (unsigned long long*)(ws + off); off += (size_t)BATCH * NANCH * 8;
  unsigned long long* wsTop  = (unsigned long long*)(ws + off); off += (size_t)BATCH * KPAD * 8;
  float* wsX1   = (float*)(ws + off); off += (size_t)BATCH * KPAD * 4;
  float* wsY1   = (float*)(ws + off); off += (size_t)BATCH * KPAD * 4;
  float* wsX2   = (float*)(ws + off); off += (size_t)BATCH * KPAD * 4;
  float* wsY2   = (float*)(ws + off); off += (size_t)BATCH * KPAD * 4;
  float* wsAr   = (float*)(ws + off); off += (size_t)BATCH * KPAD * 4;
  float* wsObj  = (float*)(ws + off); off += (size_t)BATCH * KPAD * 4;
  float* wsConf = (float*)(ws + off); off += (size_t)BATCH * KPAD * 4;
  int*   wsCls  = (int*)(ws + off);   off += (size_t)BATCH * KPAD * 4;
  unsigned* wsValid = (unsigned*)(ws + off); off += (size_t)BATCH * 32 * 4;
  unsigned* wsSup   = (unsigned*)(ws + off); off += (size_t)BATCH * KPAD * 32 * 4;

  int blocks1 = (BATCH * NANCH + 255) / 256;
  score_keys_kernel<<<blocks1, 256, 0, stream>>>(v3, wsKeys);

  select_sort_kernel<<<BATCH, 256, 0, stream>>>(wsKeys, wsTop);

  gather_kernel<<<BATCH, 256, 0, stream>>>(v3, wsTop, wsX1, wsY1, wsX2, wsY2,
                                           wsAr, wsObj, wsConf, wsCls, wsValid);

  suppress_kernel<<<BATCH * 64, 256, 0, stream>>>(wsX1, wsY1, wsX2, wsY2,
                                                  wsAr, wsCls, wsSup);

  size_t nmsLds = (size_t)(TOPK * 32 + 32) * sizeof(unsigned);
  nms_out_kernel<<<BATCH, 1024, nmsLds, stream>>>(wsSup, wsValid, wsX1, wsY1, wsX2, wsY2,
                                                  wsObj, wsConf, wsCls, out);
}
